// LSTMAttentionModel_27419071217735
// MI455X (gfx1250) — compile-verified, matrix operand order checked by reading, not run on hardware
//
#include <hip/hip_runtime.h>
#include <hip/hip_bf16.h>

// Problem constants (match reference): B=16, S=1024, D=64, H=256, L=2
#define BB 16
#define SS 1024
#define DD 64
#define HH 256
#define G4H 1024   // 4*H
#define MTOT (BB*SS)  // 16384

typedef __attribute__((ext_vector_type(16))) _Float16 v16h;
typedef __attribute__((ext_vector_type(8)))  _Float16 v8h;
typedef __attribute__((ext_vector_type(8)))  float    v8f;
typedef int v4i_b128 __attribute__((vector_size(16)));  // b128 payload type

// CDNA5 async global->LDS path (ASYNCcnt, CDNA5 ISA).
// Toolchain-probed signature: (v4i AS1*, v4i AS3*, imm offset, imm cpol).
#if defined(__has_builtin)
#if __has_builtin(__builtin_amdgcn_global_load_async_to_lds_b128)
#define HAVE_ASYNC_LDS 1
#endif
#endif

__device__ __forceinline__ void async_wait_all() {
#if defined(HAVE_ASYNC_LDS)
#if __has_builtin(__builtin_amdgcn_s_wait_asynccnt)
  __builtin_amdgcn_s_wait_asynccnt(0);
#else
  asm volatile("s_wait_asynccnt 0" ::: "memory");
#endif
#endif
}

// --------------------------------------------------------------------------
// WMMA helpers.
//
// 16x16x32 f16 WMMA, wave32. Per CDNA5 ISA §7.12.2:
//  A (16x32): lane l holds row M=l&15; half-elements e<8 -> K = khalf+e,
//             e>=8 -> K = 16+khalf+(e-8), khalf = (l>=16)*8.
//  B (32x16): mirrored with N = l&15 from the lane, same K mapping.
//  C/D f32:   element v -> row M = v + 8*(l>=16), col N = l&15.
// So a tile operand is two contiguous 8x f16 (16B) loads at kbase and kbase+16.
// --------------------------------------------------------------------------
__device__ __forceinline__ v16h ldtile(const _Float16* rowptr, int kbase) {
  union { v16h v; v8h h[2]; } u;
  u.h[0] = *(const v8h*)(rowptr + kbase);
  u.h[1] = *(const v8h*)(rowptr + kbase + 16);
  return u.v;
}

__device__ __forceinline__ v8f wmma32(v16h a, v16h b, v8f c) {
  return __builtin_amdgcn_wmma_f32_16x16x32_f16(
      /*neg_a=*/false, a, /*neg_b=*/false, b,
      /*c_mod=*/(short)0, c, /*reuse_a=*/false, /*reuse_b=*/false);
}

__device__ __forceinline__ float sigf(float x) {
  return 1.0f / (1.0f + __expf(-x));
}

// --------------------------------------------------------------------------
// Small utility kernels
// --------------------------------------------------------------------------
__global__ void cvt_f32_to_f16_kernel(const float* __restrict__ in,
                                      _Float16* __restrict__ out, int n) {
  int i = blockIdx.x * blockDim.x + threadIdx.x;
  if (i < n) out[i] = (_Float16)in[i];
}

__global__ void bias_sum_kernel(const float* __restrict__ a,
                                const float* __restrict__ b,
                                float* __restrict__ o, int n) {
  int i = blockIdx.x * blockDim.x + threadIdx.x;
  if (i < n) o[i] = a[i] + b[i];
}

__global__ void mask_out_kernel(float* __restrict__ out,
                                const int* __restrict__ lengths) {
  int i = blockIdx.x * blockDim.x + threadIdx.x;
  if (i >= MTOT * DD) return;
  int row = i / DD;
  int s = row & (SS - 1);
  int b = row >> 10;  // /S
  if (s >= lengths[b]) out[i] = 0.0f;
}

// --------------------------------------------------------------------------
// Generic f16 WMMA GEMM:  Out[M,N] = A[M,K] @ W[N,K]^T + bias[N]
// A row-major [M,K], W row-major [N,K], bias f32 per column.
// Block: 256 threads = 8 waves; block tile 128x64. All 8 waves consume the
// SAME 64 B-columns, so the B panel is staged through LDS once per block:
// double-buffered 32(K)x64(N) f16 chunks, filled with async global->LDS
// loads (one b128 per thread per chunk) overlapped with WMMA on the other
// buffer. A-operand is register double-buffered.
// mode: 0 = f16 row-major out, 1 = f16 transposed out (Out[N, Mtot], ldo=Mtot),
//       2 = f32 row-major out.
// --------------------------------------------------------------------------
#define GM_F16  0
#define GM_F16T 1
#define GM_F32  2

__global__ __launch_bounds__(256) void gemm_wmma_kernel(
    const _Float16* __restrict__ A, int lda,
    const _Float16* __restrict__ W, int ldb,
    const float* __restrict__ bias,
    void* __restrict__ Out, int ldo,
    int M, int N, int K, int mode) {
  __shared__ _Float16 bstage[2][64 * 32];  // 2 x 4KB B-panel chunks [n][k]

  const int lane  = threadIdx.x & 31;
  const int wave  = threadIdx.x >> 5;
  const int l15   = lane & 15;
  const int hi    = lane >> 4;
  const int khalf = hi * 8;
  const int rb = blockIdx.y * 128 + wave * 16;
  const int cb = blockIdx.x * 64;
  const int arow = rb + l15;

  // B-panel stager: thread tid moves one 16B chunk per K-chunk.
  const int tn = threadIdx.x >> 2;         // 0..63: panel row (N)
  const int tk = (threadIdx.x & 3) * 8;    // 0,8,16,24: K offset
  const _Float16* gsrc = W + (size_t)(cb + tn) * ldb + tk;
  _Float16* lds_dst0 = &bstage[0][tn * 32 + tk];
  _Float16* lds_dst1 = &bstage[1][tn * 32 + tk];

  auto stage = [&](int buf, int kk) {
    _Float16* ld = buf ? lds_dst1 : lds_dst0;
#if defined(HAVE_ASYNC_LDS)
    __builtin_amdgcn_global_load_async_to_lds_b128(
        (__attribute__((address_space(1))) v4i_b128*)(gsrc + kk),
        (__attribute__((address_space(3))) v4i_b128*)ld,
        /*offset=*/0, /*cpol=*/0);
#else
    *(v8h*)ld = *(const v8h*)(gsrc + kk);
#endif
  };

  const int nk = K >> 5;  // K chunks of 32
  stage(0, 0);
  async_wait_all();
  __syncthreads();

  v8f acc[4] = {};
  v16h a = ldtile(A + (size_t)arow * lda, khalf);

  for (int kc = 0; kc < nk; ++kc) {
    const int cur = kc & 1;
    const bool more = (kc + 1 < nk);
    if (more) stage(cur ^ 1, (kc + 1) * 32);           // async fill next chunk
    v16h an = a;
    if (more) an = ldtile(A + (size_t)arow * lda, (kc + 1) * 32 + khalf);
#pragma unroll
    for (int nt = 0; nt < 4; ++nt) {
      v16h b = ldtile(&bstage[cur][(nt * 16 + l15) * 32], khalf);
      acc[nt] = wmma32(a, b, acc[nt]);
    }
    a = an;
    async_wait_all();   // our async fills for the next buffer have landed
    __syncthreads();    // everyone done reading cur, next buffer visible
  }

#pragma unroll
  for (int nt = 0; nt < 4; ++nt) {
#pragma unroll
    for (int v = 0; v < 8; ++v) {
      int r = rb + v + hi * 8;
      int c = cb + nt * 16 + l15;
      float val = acc[nt][v] + bias[c];
      if (mode == GM_F16)
        ((_Float16*)Out)[(size_t)r * ldo + c] = (_Float16)val;
      else if (mode == GM_F16T)
        ((_Float16*)Out)[(size_t)c * ldo + r] = (_Float16)val;
      else
        ((float*)Out)[(size_t)r * ldo + c] = val;
    }
  }
}

// --------------------------------------------------------------------------
// LSTM recurrence. Single persistent workgroup: 1024 threads = 32 waves.
// gx [B*S, 4H] f16 already contains x@W_ih^T + (b_ih+b_hh).
// whh [4H, H] f16 streamed from L2 (each wave's 32 gate columns are unique,
// so there is no cross-wave reuse to stage). h kept in LDS f16; per-step
// gates staged in LDS f16; c-state lives in registers (fixed thread->(b,j)
// mapping across steps). hout zeroed at padded positions (pad_packed
// semantics) while the recurrence itself continues, like the reference.
// --------------------------------------------------------------------------
__global__ __launch_bounds__(1024) void lstm_kernel(
    const _Float16* __restrict__ gx,
    const _Float16* __restrict__ whh,
    const int* __restrict__ lengths,
    _Float16* __restrict__ hout) {
  __shared__ _Float16 gbuf[BB * G4H];   // 32 KB: per-step gates (f16)
  __shared__ _Float16 hbuf[BB * HH];    // 8 KB:  h_{t-1}

  const int tid   = threadIdx.x;
  const int lane  = tid & 31;
  const int wave  = tid >> 5;
  const int l15   = lane & 15;
  const int hi    = lane >> 4;
  const int khalf = hi * 8;
  const int cb    = wave * 32;          // this wave's 32 gate columns

  const int e0   = tid * 4;             // phase-2: 4 consecutive (b,j) cells
  const int brow = e0 >> 8;
  const int j0   = e0 & 255;
  const int mylen = lengths[brow];
  float cst[4] = {0.f, 0.f, 0.f, 0.f};

  for (int i = tid; i < BB * HH; i += 1024) hbuf[i] = (_Float16)0.f;
  __syncthreads();

  for (int t = 0; t < SS; ++t) {
    // ---- phase 1: g = gx[:,t,:] + h @ W_hh^T  (16 WMMAs per wave) ----
    v8f acc0 = {}, acc1 = {};
#pragma unroll
    for (int v = 0; v < 8; ++v) {
      int b = v + hi * 8;
      size_t base = ((size_t)b * SS + (size_t)t) * G4H;
      acc0[v] = (float)gx[base + cb + l15];
      acc1[v] = (float)gx[base + cb + 16 + l15];
    }
#pragma unroll
    for (int kk = 0; kk < HH; kk += 32) {
      v16h a  = ldtile(hbuf + l15 * HH, kk + khalf);
      v16h b0 = ldtile(whh + (size_t)(cb + l15) * HH, kk + khalf);
      v16h b1 = ldtile(whh + (size_t)(cb + 16 + l15) * HH, kk + khalf);
      acc0 = wmma32(a, b0, acc0);
      acc1 = wmma32(a, b1, acc1);
    }
#pragma unroll
    for (int v = 0; v < 8; ++v) {
      int b = v + hi * 8;
      gbuf[b * G4H + cb + l15]      = (_Float16)acc0[v];
      gbuf[b * G4H + cb + 16 + l15] = (_Float16)acc1[v];
    }
    __syncthreads();

    // ---- phase 2: gate nonlinearity + state update ----
    const bool valid = (t < mylen);
#pragma unroll
    for (int u = 0; u < 4; ++u) {
      int j = j0 + u;
      float ig = (float)gbuf[brow * G4H + j];
      float fg = (float)gbuf[brow * G4H + HH + j];
      float gg = (float)gbuf[brow * G4H + 2 * HH + j];
      float og = (float)gbuf[brow * G4H + 3 * HH + j];
      float c = sigf(fg) * cst[u] + sigf(ig) * tanhf(gg);
      float h = sigf(og) * tanhf(c);
      cst[u] = c;
      hbuf[brow * HH + j] = (_Float16)h;
      hout[((size_t)brow * SS + t) * HH + j] = valid ? (_Float16)h : (_Float16)0.f;
    }
    __syncthreads();
  }
}

// --------------------------------------------------------------------------
// Causal flash attention (no 1/sqrt(d) scaling, matching the reference).
// Grid: (S/64, B); 128 threads = 4 waves; wave w owns queries q0..q0+15,
// fully independent -> per-wave online softmax in registers.
// Q,K row-major [B*S, H] f16; Vt transposed [H, B*S] f16 so the P@V
// B-operand is contiguous. P is transposed to the WMMA A-layout via a
// per-wave LDS bounce (same-wave DS ops are in-order via DScnt).
// --------------------------------------------------------------------------
__global__ __launch_bounds__(128) void attn_kernel(
    const _Float16* __restrict__ Q,
    const _Float16* __restrict__ Km,
    const _Float16* __restrict__ Vt,
    _Float16* __restrict__ Out) {
  __shared__ _Float16 pbuf[4][16 * 32];

  const int tid   = threadIdx.x;
  const int lane  = tid & 31;
  const int wave  = tid >> 5;
  const int l15   = lane & 15;
  const int hi    = lane >> 4;
  const int khalf = hi * 8;
  const int b     = blockIdx.y;
  const int q0    = blockIdx.x * 64 + wave * 16;   // query base within batch
  const size_t gq = (size_t)b * SS + q0;           // global query row base

  // Preload all Q A-tiles for this wave's 16 query rows (K = 256 -> 8 tiles).
  v16h aQ[8];
#pragma unroll
  for (int kk = 0; kk < 8; ++kk)
    aQ[kk] = ldtile(Q + (gq + l15) * HH, kk * 32 + khalf);

  v8f o[16] = {};
  float mrow[8], lrow[8];
#pragma unroll
  for (int v = 0; v < 8; ++v) { mrow[v] = -3.0e38f; lrow[v] = 0.f; }

  _Float16* pw = &pbuf[wave][0];
  const int nkey = q0 + 16;  // keys needed: 0..q0+15 (causal)

  for (int kb = 0; kb < nkey; kb += 32) {
    const size_t gk = (size_t)b * SS + kb;

    // ---- scores: S = Q @ K^T for 32 keys (16 WMMAs) ----
    v8f s0 = {}, s1 = {};
#pragma unroll
    for (int kk = 0; kk < 8; ++kk) {
      v16h bk0 = ldtile(Km + (gk + l15) * HH, kk * 32 + khalf);
      v16h bk1 = ldtile(Km + (gk + 16 + l15) * HH, kk * 32 + khalf);
      s0 = wmma32(aQ[kk], bk0, s0);
      s1 = wmma32(aQ[kk], bk1, s1);
    }

    // ---- online softmax (per query row; 16-lane shuffle reductions) ----
    float scl[8];
#pragma unroll
    for (int v = 0; v < 8; ++v) {
      const int qrow = q0 + v + hi * 8;
      const int k0 = kb + l15, k1 = kb + 16 + l15;
      float sv0 = (k0 <= qrow) ? s0[v] : -3.0e38f;
      float sv1 = (k1 <= qrow) ? s1[v] : -3.0e38f;
      float mx = fmaxf(sv0, sv1);
#pragma unroll
      for (int d = 1; d < 16; d <<= 1) mx = fmaxf(mx, __shfl_xor(mx, d, 32));
      float mnew  = fmaxf(mrow[v], mx);
      float scale = __expf(mrow[v] - mnew);
      float p0 = __expf(sv0 - mnew);
      float p1 = __expf(sv1 - mnew);
      float ps = p0 + p1;
#pragma unroll
      for (int d = 1; d < 16; d <<= 1) ps += __shfl_xor(ps, d, 32);
      lrow[v] = lrow[v] * scale + ps;
      mrow[v] = mnew;
      scl[v]  = scale;
      const int prow = v + hi * 8;  // C-layout row -> LDS row
      pw[prow * 32 + l15]      = (_Float16)p0;
      pw[prow * 32 + 16 + l15] = (_Float16)p1;
    }

    // rescale accumulated O (scale has the same C-layout as o)
#pragma unroll
    for (int nt = 0; nt < 16; ++nt)
#pragma unroll
      for (int v = 0; v < 8; ++v) o[nt][v] *= scl[v];

    // Same-wave DS ops are in-order (DScnt); just stop compiler reordering.
    asm volatile("" ::: "memory");

    // ---- O += P @ V  (16 WMMAs; A from LDS bounce, B from Vt contiguous) ----
    v16h ap = ldtile(pw + l15 * 32, khalf);
#pragma unroll
    for (int nt = 0; nt < 16; ++nt) {
      v16h bv = ldtile(Vt + (size_t)(nt * 16 + l15) * MTOT + gk, khalf);
      o[nt] = wmma32(ap, bv, o[nt]);
    }
  }

  // ---- epilogue: O /= l, store f16 ----
  float linv[8];
#pragma unroll
  for (int v = 0; v < 8; ++v) linv[v] = 1.0f / lrow[v];
#pragma unroll
  for (int nt = 0; nt < 16; ++nt)
#pragma unroll
    for (int v = 0; v < 8; ++v) {
      size_t r = gq + v + hi * 8;
      Out[r * HH + nt * 16 + l15] = (_Float16)(o[nt][v] * linv[v]);
    }
}

// --------------------------------------------------------------------------
// Host launcher
// --------------------------------------------------------------------------
extern "C" void kernel_launch(void* const* d_in, const int* in_sizes, int n_in,
                              void* d_out, int out_size, void* d_ws, size_t ws_size,
                              hipStream_t stream) {
  const float* x     = (const float*)d_in[0];   // [B,S,D]
  const int*   lens  = (const int*)  d_in[1];   // [B]
  const float* w_ih  = (const float*)d_in[2];   // [4H,D]
  const float* w_hh  = (const float*)d_in[3];   // [4H,H]
  const float* b_ih  = (const float*)d_in[4];   // [4H]
  const float* b_hh  = (const float*)d_in[5];   // [4H]
  const float* Wq    = (const float*)d_in[6];   // [L,H,H]
  const float* bq    = (const float*)d_in[7];   // [L,H]
  const float* Wk    = (const float*)d_in[8];
  const float* bk    = (const float*)d_in[9];
  const float* Wv    = (const float*)d_in[10];
  const float* bv    = (const float*)d_in[11];
  const float* Wp    = (const float*)d_in[12];  // [D,H]
  const float* bp    = (const float*)d_in[13];  // [D]
  float* out = (float*)d_out;                   // [B,S,D]

  // Workspace carve-up (256B aligned)
  char* ws = (char*)d_ws;
  size_t off = 0;
  auto carve = [&](size_t bytes) {
    char* p = ws + off;
    off += (bytes + 255) & ~(size_t)255;
    return p;
  };
  _Float16* x16     = (_Float16*)carve((size_t)MTOT * DD * 2);
  _Float16* wih16   = (_Float16*)carve((size_t)G4H * DD * 2);
  _Float16* whh16   = (_Float16*)carve((size_t)G4H * HH * 2);
  _Float16* wq16    = (_Float16*)carve((size_t)2 * HH * HH * 2);
  _Float16* wk16    = (_Float16*)carve((size_t)2 * HH * HH * 2);
  _Float16* wv16    = (_Float16*)carve((size_t)2 * HH * HH * 2);
  _Float16* wp16    = (_Float16*)carve((size_t)DD * HH * 2);
  float*    biasSum = (float*)   carve((size_t)G4H * 4);
  _Float16* gates16 = (_Float16*)carve((size_t)MTOT * G4H * 2);
  _Float16* hA      = (_Float16*)carve((size_t)MTOT * HH * 2);
  _Float16* hB      = (_Float16*)carve((size_t)MTOT * HH * 2);
  _Float16* Q16     = (_Float16*)carve((size_t)MTOT * HH * 2);
  _Float16* K16     = (_Float16*)carve((size_t)MTOT * HH * 2);
  _Float16* Vt16    = (_Float16*)carve((size_t)MTOT * HH * 2);  // [H, B*S]
  (void)ws_size; (void)n_in; (void)in_sizes; (void)out_size;

  auto cvt = [&](const float* src, _Float16* dst, int n) {
    cvt_f32_to_f16_kernel<<<(n + 255) / 256, 256, 0, stream>>>(src, dst, n);
  };
  cvt(x,    x16,   MTOT * DD);
  cvt(w_ih, wih16, G4H * DD);
  cvt(w_hh, whh16, G4H * HH);
  cvt(Wq,   wq16,  2 * HH * HH);
  cvt(Wk,   wk16,  2 * HH * HH);
  cvt(Wv,   wv16,  2 * HH * HH);
  cvt(Wp,   wp16,  DD * HH);
  bias_sum_kernel<<<(G4H + 255) / 256, 256, 0, stream>>>(b_ih, b_hh, biasSum, G4H);

  // gates_x = x @ W_ih^T + (b_ih + b_hh):  [16384,64] x [64,1024]
  gemm_wmma_kernel<<<dim3(G4H / 64, MTOT / 128), 256, 0, stream>>>(
      x16, DD, wih16, DD, biasSum, gates16, G4H, MTOT, G4H, DD, GM_F16);

  // LSTM recurrence (single persistent workgroup, 32 waves)
  lstm_kernel<<<1, 1024, 0, stream>>>(gates16, whh16, lens, hA);

  // Two causal attention layers
  _Float16* cur = hA;
  _Float16* nxt = hB;
  for (int l = 0; l < 2; ++l) {
    const _Float16* wq_l = wq16 + (size_t)l * HH * HH;
    const _Float16* wk_l = wk16 + (size_t)l * HH * HH;
    const _Float16* wv_l = wv16 + (size_t)l * HH * HH;
    gemm_wmma_kernel<<<dim3(HH / 64, MTOT / 128), 256, 0, stream>>>(
        cur, HH, wq_l, HH, bq + l * HH, Q16, HH, MTOT, HH, HH, GM_F16);
    gemm_wmma_kernel<<<dim3(HH / 64, MTOT / 128), 256, 0, stream>>>(
        cur, HH, wk_l, HH, bk + l * HH, K16, HH, MTOT, HH, HH, GM_F16);
    // V stored transposed [H, B*S] for contiguous P@V B-operand loads
    gemm_wmma_kernel<<<dim3(HH / 64, MTOT / 128), 256, 0, stream>>>(
        cur, HH, wv_l, HH, bv + l * HH, Vt16, MTOT, MTOT, HH, HH, GM_F16T);
    attn_kernel<<<dim3(SS / 64, BB), 128, 0, stream>>>(Q16, K16, Vt16, nxt);
    _Float16* t = cur; cur = nxt; nxt = t;
  }

  // Final projection -> f32 d_out, then zero padded rows
  gemm_wmma_kernel<<<dim3(DD / 64, MTOT / 128), 256, 0, stream>>>(
      cur, HH, wp16, HH, bp, out, DD, MTOT, DD, HH, GM_F32);
  mask_out_kernel<<<(MTOT * DD + 255) / 256, 256, 0, stream>>>(out, lens);
}
